// MultiHeadAttentionKernel_65481071402723
// MI455X (gfx1250) — matrix operand order repaired and checked by reading, not compiled, not run on hardware
//
#include <hip/hip_runtime.h>
#include <stdint.h>

// ---------- types ----------
typedef __attribute__((ext_vector_type(16))) __bf16 v16bf;
typedef __attribute__((ext_vector_type(8)))  __bf16 v8bf;
typedef __attribute__((ext_vector_type(4)))  __bf16 v4bf;
typedef __attribute__((ext_vector_type(8)))  float  v8f;
typedef __attribute__((ext_vector_type(4)))  float  v4f;

union bfpack { v16bf v; v8bf h[2]; };

__device__ inline v16bf pack16(v8bf lo, v8bf hi) {
  bfpack u; u.h[0] = lo; u.h[1] = hi; return u.v;
}

// fp32 -> bf16 round-to-nearest-even
__device__ inline __bf16 f2bf(float f) {
  union { float f; uint32_t u; } v; v.f = f;
  uint32_t r = v.u + 0x7FFFu + ((v.u >> 16) & 1u);
  uint16_t h = (uint16_t)(r >> 16);
  __bf16 o; __builtin_memcpy(&o, &h, 2);
  return o;
}
__device__ inline float bf2f(__bf16 b) {
  uint16_t h; __builtin_memcpy(&h, &b, 2);
  union { uint32_t u; float f; } v; v.u = ((uint32_t)h) << 16;
  return v.f;
}
__device__ inline uint32_t packpair(__bf16 a, __bf16 b) {
  uint16_t x, y; __builtin_memcpy(&x, &a, 2); __builtin_memcpy(&y, &b, 2);
  return (uint32_t)x | ((uint32_t)y << 16);
}

// ---------- CDNA5 async global->LDS copy (per-lane 16B), ASYNCcnt tracked ----------
__device__ inline void async_copy_b128(void* lds_ptr, const void* gaddr) {
  // generic shared-pointer low 32 bits == wave-relative LDS byte offset (ISA 10.2)
  uint32_t lds_off = (uint32_t)(uintptr_t)lds_ptr;
  asm volatile("global_load_async_to_lds_b128 %0, %1, off"
               :: "v"(lds_off), "v"((uint64_t)(uintptr_t)gaddr)
               : "memory");
}
__device__ inline void wait_async0() {
#if __has_builtin(__builtin_amdgcn_s_wait_asynccnt)
  __builtin_amdgcn_s_wait_asynccnt(0);
#else
  asm volatile("s_wait_asynccnt 0x0" ::: "memory");
#endif
}

// ---------- problem constants ----------
#define BQ   4
#define SEQ  2048
#define EMB  1024
#define NH   16
#define DK   64
#define MROWS (BQ * SEQ)          // 8192

// ---------- workspace layout (bytes) ----------
#define OFF_XB  ((size_t)0)
#define OFF_WQ  (OFF_XB + (size_t)MROWS * EMB * 2)
#define OFF_WK  (OFF_WQ + (size_t)EMB * EMB * 2)
#define OFF_WV  (OFF_WK + (size_t)EMB * EMB * 2)
#define OFF_WO  (OFF_WV + (size_t)EMB * EMB * 2)
#define OFF_Q   (OFF_WO + (size_t)EMB * EMB * 2)
#define OFF_K   (OFF_Q  + (size_t)MROWS * EMB * 2)
#define OFF_V   (OFF_K  + (size_t)MROWS * EMB * 2)
#define OFF_AO  (OFF_V  + (size_t)MROWS * EMB * 2)
#define OFF_QN  (OFF_AO + (size_t)MROWS * EMB * 2)
#define OFF_KN  (OFF_QN + (size_t)BQ * NH * SEQ * 4)

// ---------- kernel 1: fp32 -> bf16 convert (vectorized x4) ----------
__global__ void cvt_f32_bf16(const float* __restrict__ src, __bf16* __restrict__ dst, int n4) {
  for (int i = blockIdx.x * blockDim.x + threadIdx.x; i < n4; i += blockDim.x * gridDim.x) {
    v4f f = ((const v4f*)src)[i];
    v4bf o;
#pragma unroll
    for (int j = 0; j < 4; ++j) o[j] = f2bf(f[j]);
    ((v4bf*)dst)[i] = o;
  }
}

// ---------- kernel 2/5: C = A[M,K] * W[N,K]^T, async-LDS staged, ping-pong ----------
// MODE 0: store bf16 in [B,H,S,DK] head layout (q/k/v); z selects weight/out.
// MODE 1: store fp32 plain [M,N] to d_out.
template <int MODE>
__global__ __launch_bounds__(256) void gemm_wt(
    const __bf16* __restrict__ A,
    const __bf16* __restrict__ W0, const __bf16* __restrict__ W1, const __bf16* __restrict__ W2,
    void* __restrict__ O0, void* __restrict__ O1, void* __restrict__ O2) {
  const int K = EMB;
  const int tid  = threadIdx.x;
  const int lane = tid & 31;
  const int wave = tid >> 5;
  const int mw = wave >> 2;                       // 0..1
  const int nw = wave & 3;                        // 0..3
  const int row0l  = mw * 64;
  const int ncol0l = nw * 32;
  const int m16 = lane & 15, half = lane >> 4;

  const __bf16* W = (blockIdx.z == 0) ? W0 : (blockIdx.z == 1) ? W1 : W2;
  void*         O = (blockIdx.z == 0) ? O0 : (blockIdx.z == 1) ? O1 : O2;

  // 128x32 tiles, rows padded to 40 elements (80B stride -> conflict-free)
  __shared__ __bf16 As[2][128][40];
  __shared__ __bf16 Ws[2][128][40];

  // per-thread staging: 2 chunks of 8 bf16 for A, 2 for W (one async b128 each)
  const int c0 = tid, c1 = tid + 256;
  const int rA0 = c0 >> 2, sA0 = c0 & 3;
  const int rA1 = c1 >> 2, sA1 = c1 & 3;
  const __bf16* gA0 = A + (size_t)(blockIdx.x * 128 + rA0) * EMB + sA0 * 8;
  const __bf16* gA1 = A + (size_t)(blockIdx.x * 128 + rA1) * EMB + sA1 * 8;
  const __bf16* gW0 = W + (size_t)(blockIdx.y * 128 + rA0) * EMB + sA0 * 8;
  const __bf16* gW1 = W + (size_t)(blockIdx.y * 128 + rA1) * EMB + sA1 * 8;

  v8f acc[4][2] = {};

  auto issueAsync = [&](int nb, int k) {
    async_copy_b128(&As[nb][rA0][sA0 * 8], gA0 + k);
    async_copy_b128(&As[nb][rA1][sA1 * 8], gA1 + k);
    async_copy_b128(&Ws[nb][rA0][sA0 * 8], gW0 + k);
    async_copy_b128(&Ws[nb][rA1][sA1 * 8], gW1 + k);
  };
  auto compute = [&](int buf) {
    const __bf16* bp0 = &Ws[buf][ncol0l + m16][half * 8];
    const __bf16* bp1 = &Ws[buf][ncol0l + 16 + m16][half * 8];
    v16bf b0 = pack16(*(const v8bf*)bp0, *(const v8bf*)(bp0 + 16));
    v16bf b1 = pack16(*(const v8bf*)bp1, *(const v8bf*)(bp1 + 16));
#pragma unroll
    for (int mi = 0; mi < 4; ++mi) {
      const __bf16* ap = &As[buf][row0l + mi * 16 + m16][half * 8];
      v16bf a = pack16(*(const v8bf*)ap, *(const v8bf*)(ap + 16));
      acc[mi][0] = __builtin_amdgcn_wmma_f32_16x16x32_bf16(false, a, false, b0, (short)0, acc[mi][0], false, false);
      acc[mi][1] = __builtin_amdgcn_wmma_f32_16x16x32_bf16(false, a, false, b1, (short)0, acc[mi][1], false, false);
    }
  };

  // prologue: stream k-step 0 directly into LDS, bypassing VGPRs
  issueAsync(0, 0);

  // explicit ping-pong: compile-time buffer indices, one barrier per phase.
  // async writes into buffer b are only issued after the barrier of the phase
  // that computed from b; each wave drains its own ASYNCcnt before the barrier.
  for (int k0 = 0; k0 < K; k0 += 64) {
    // phase 0: compute buf0 @ k0, stream buf1 @ k0+32
    wait_async0();
    __syncthreads();
    issueAsync(1, k0 + 32);                  // k0+32 < K always (K % 64 == 0)
    if (k0 + 96 < K) {                       // L2 prefetch two steps ahead
      __builtin_prefetch(gA0 + k0 + 96, 0, 0);
      __builtin_prefetch(gW0 + k0 + 96, 0, 0);
    }
    compute(0);
    // phase 1: compute buf1 @ k0+32, stream buf0 @ k0+64
    wait_async0();
    __syncthreads();
    if (k0 + 64 < K) issueAsync(0, k0 + 64);
    compute(1);
  }

#pragma unroll
  for (int mi = 0; mi < 4; ++mi) {
#pragma unroll
    for (int ni = 0; ni < 2; ++ni) {
#pragma unroll
      for (int r = 0; r < 8; ++r) {
        int m = blockIdx.x * 128 + row0l + mi * 16 + r + half * 8;
        int n = blockIdx.y * 128 + ncol0l + ni * 16 + m16;
        if (MODE == 0) {
          int b = m >> 11, s = m & (SEQ - 1);
          int h = n >> 6,  d = n & (DK - 1);
          ((__bf16*)O)[(((size_t)b * NH + h) * SEQ + s) * DK + d] = f2bf(acc[mi][ni][r]);
        } else {
          ((float*)O)[(size_t)m * EMB + n] = acc[mi][ni][r];
        }
      }
    }
  }
}

// ---------- kernel 3: row norms of bf16 [rows, DK] ----------
__global__ void norm_kernel(const __bf16* __restrict__ X, float* __restrict__ N, int rows) {
  int r = blockIdx.x * blockDim.x + threadIdx.x;
  if (r >= rows) return;
  const __bf16* p = X + (size_t)r * DK;
  float acc = 0.f;
#pragma unroll
  for (int c = 0; c < 8; ++c) {
    v8bf v = *(const v8bf*)(p + c * 8);
#pragma unroll
    for (int j = 0; j < 8; ++j) { float f = bf2f(v[j]); acc += f * f; }
  }
  N[r] = acc;
}

// ---------- kernel 4: fused RBF attention, async-staged K, ping-pong ----------
__global__ __launch_bounds__(256) void attn_kernel(
    const __bf16* __restrict__ Qb, const __bf16* __restrict__ Kb, const __bf16* __restrict__ Vb,
    const float* __restrict__ Qn, const float* __restrict__ Kn,
    const int* __restrict__ Mask,
    __bf16* __restrict__ Ob) {  // [B,S,E] bf16
  const int bh = blockIdx.y;          // 0..63
  const int b  = bh >> 4;
  const int h  = bh & 15;
  const int tid = threadIdx.x, lane = tid & 31, wave = tid >> 5;
  const int m16 = lane & 15, half = lane >> 4;
  const int rowbase = blockIdx.x * 128 + wave * 16;   // 16 query rows / wave

  __shared__ __bf16 Ks[2][32][72];     // K chunk, row-major [key][dk], padded
  __shared__ __bf16 Vt[2][DK][40];     // V^T chunk [d][key], padded
  __shared__ float  knEff[2][32];      // key norms, +3e38 when masked
  __shared__ __bf16 pT[8][16][40];     // per-wave P tile [row][key]

  const size_t headoff = (size_t)bh * SEQ * DK;
  const __bf16* Qh = Qb + headoff;
  const __bf16* Kh = Kb + headoff;
  const __bf16* Vh = Vb + headoff;

  // ---- hoisted per-wave Q fragments and q-norms ----
  v16bf qa[2];
#pragma unroll
  for (int c = 0; c < 2; ++c) {
    const __bf16* p = Qh + (size_t)(rowbase + m16) * DK + c * 32 + half * 8;
    qa[c] = pack16(*(const v8bf*)p, *(const v8bf*)(p + 16));
  }
  float qn_l[8];
#pragma unroll
  for (int r = 0; r < 8; ++r)
    qn_l[r] = Qn[(size_t)bh * SEQ + rowbase + r + half * 8];

  // ---- per-thread staging assignments ----
  const int kKey = tid >> 3, kSeg = tid & 7;          // K: 32 keys x 8 segs of 8 (async b128)
  const int vKp  = tid >> 4, vDb  = tid & 15;         // V: 16 key-pairs x 16 d-blocks of 4
  const int vKey0 = vKp * 2, vD0 = vDb * 4;

  // prologue: async K chunk 0 straight into LDS; V/kn for chunk 0 into registers
  async_copy_b128(&Ks[0][kKey][kSeg * 8], Kh + (size_t)kKey * DK + kSeg * 8);
  v4bf rv0 = *(const v4bf*)(Vh + (size_t)vKey0 * DK + vD0);
  v4bf rv1 = *(const v4bf*)(Vh + (size_t)(vKey0 + 1) * DK + vD0);
  float rkn = 0.f;
  if (tid < 32) {
    float kn = Kn[(size_t)bh * SEQ + tid];
    rkn = Mask[b * SEQ + tid] ? kn : 3.0e38f;
  }

  v8f oacc[4] = {};

  auto phase = [&](int buf, int kb0) {
    // ---- commit transposed V + key norms; K arrives via async ----
#pragma unroll
    for (int j = 0; j < 4; ++j)
      *(uint32_t*)(&Vt[buf][vD0 + j][vKey0]) = packpair(rv0[j], rv1[j]);
    if (tid < 32) knEff[buf][tid] = rkn;
    wait_async0();            // this chunk's async K writes have landed
    __syncthreads();

    // ---- stream next chunk (async K; V/kn into registers) ----
    if (kb0 + 32 < SEQ) {
      async_copy_b128(&Ks[buf ^ 1][kKey][kSeg * 8],
                      Kh + (size_t)(kb0 + 32 + kKey) * DK + kSeg * 8);
      rv0 = *(const v4bf*)(Vh + (size_t)(kb0 + 32 + vKey0) * DK + vD0);
      rv1 = *(const v4bf*)(Vh + (size_t)(kb0 + 32 + vKey0 + 1) * DK + vD0);
      if (tid < 32) {
        float kn = Kn[(size_t)bh * SEQ + kb0 + 32 + tid];
        rkn = Mask[b * SEQ + kb0 + 32 + tid] ? kn : 3.0e38f;
      }
    }

    // ---- S = Q K^T from LDS (two 16-key tiles) ----
    v8f s0 = {}, s1 = {};
#pragma unroll
    for (int c = 0; c < 2; ++c) {
      const __bf16* kp0 = &Ks[buf][m16][c * 32 + half * 8];
      v16bf kf0 = pack16(*(const v8bf*)kp0, *(const v8bf*)(kp0 + 16));
      s0 = __builtin_amdgcn_wmma_f32_16x16x32_bf16(false, qa[c], false, kf0, (short)0, s0, false, false);
      const __bf16* kp1 = &Ks[buf][16 + m16][c * 32 + half * 8];
      v16bf kf1 = pack16(*(const v8bf*)kp1, *(const v8bf*)(kp1 + 16));
      s1 = __builtin_amdgcn_wmma_f32_16x16x32_bf16(false, qa[c], false, kf1, (short)0, s1, false, false);
    }

    // ---- P = exp(2*s - |q|^2 - |k|^2), mask folded into knEff ----
    float kn0 = knEff[buf][m16];
    float kn1 = knEff[buf][16 + m16];
#pragma unroll
    for (int r = 0; r < 8; ++r) {
      float p0 = __expf(2.0f * s0[r] - qn_l[r] - kn0);
      float p1 = __expf(2.0f * s1[r] - qn_l[r] - kn1);
      int m = r + half * 8;
      pT[wave][m][m16]      = f2bf(p0);
      pT[wave][m][16 + m16] = f2bf(p1);
    }
    // same-wave DS ops are in-order; wait + compile barrier before re-reading P
    asm volatile("s_wait_dscnt 0x0" ::: "memory");

    const __bf16* pp = &pT[wave][m16][half * 8];
    v16bf pa = pack16(*(const v8bf*)pp, *(const v8bf*)(pp + 16));

    // ---- O += P V from LDS ----
#pragma unroll
    for (int j = 0; j < 4; ++j) {
      const __bf16* vp = &Vt[buf][j * 16 + m16][half * 8];
      v16bf vf = pack16(*(const v8bf*)vp, *(const v8bf*)(vp + 16));
      oacc[j] = __builtin_amdgcn_wmma_f32_16x16x32_bf16(false, pa, false, vf, (short)0, oacc[j], false, false);
    }
  };

  // explicit ping-pong over 64 key-chunks (SEQ/32), compile-time buffer ids
  for (int kb0 = 0; kb0 < SEQ; kb0 += 64) {
    phase(0, kb0);
    phase(1, kb0 + 32);
  }

  // ---- store to [B,S,E] bf16 for the Wo GEMM ----
#pragma unroll
  for (int j = 0; j < 4; ++j) {
#pragma unroll
    for (int r = 0; r < 8; ++r) {
      int s = rowbase + r + half * 8;
      int e = h * DK + j * 16 + m16;
      Ob[((size_t)b * SEQ + s) * EMB + e] = f2bf(oacc[j][r]);
    }
  }
}

// ---------- launcher ----------
extern "C" void kernel_launch(void* const* d_in, const int* in_sizes, int n_in,
                              void* d_out, int out_size, void* d_ws, size_t ws_size,
                              hipStream_t stream) {
  const float* x    = (const float*)d_in[0];
  const int*   mask = (const int*)d_in[1];
  const float* Wq   = (const float*)d_in[2];
  const float* Wk   = (const float*)d_in[3];
  const float* Wv   = (const float*)d_in[4];
  const float* Wo   = (const float*)d_in[5];

  char* ws = (char*)d_ws;
  __bf16* xb  = (__bf16*)(ws + OFF_XB);
  __bf16* wqb = (__bf16*)(ws + OFF_WQ);
  __bf16* wkb = (__bf16*)(ws + OFF_WK);
  __bf16* wvb = (__bf16*)(ws + OFF_WV);
  __bf16* wob = (__bf16*)(ws + OFF_WO);
  __bf16* qb  = (__bf16*)(ws + OFF_Q);
  __bf16* kb  = (__bf16*)(ws + OFF_K);
  __bf16* vb  = (__bf16*)(ws + OFF_V);
  __bf16* aob = (__bf16*)(ws + OFF_AO);
  float*  qn  = (float*)(ws + OFF_QN);
  float*  kn  = (float*)(ws + OFF_KN);

  // 1) convert operands to bf16
  cvt_f32_bf16<<<2048, 256, 0, stream>>>(x,  xb,  MROWS * EMB / 4);
  cvt_f32_bf16<<<512,  256, 0, stream>>>(Wq, wqb, EMB * EMB / 4);
  cvt_f32_bf16<<<512,  256, 0, stream>>>(Wk, wkb, EMB * EMB / 4);
  cvt_f32_bf16<<<512,  256, 0, stream>>>(Wv, wvb, EMB * EMB / 4);
  cvt_f32_bf16<<<512,  256, 0, stream>>>(Wo, wob, EMB * EMB / 4);

  // 2) Q/K/V projections (one batched WMMA GEMM, z selects weight)
  gemm_wt<0><<<dim3(MROWS / 128, EMB / 128, 3), 256, 0, stream>>>(
      xb, wqb, wkb, wvb, qb, kb, vb);

  // 3) squared norms (from bf16 values, consistent with the WMMA Gram matrix)
  norm_kernel<<<(BQ * NH * SEQ) / 256, 256, 0, stream>>>(qb, qn, BQ * NH * SEQ);
  norm_kernel<<<(BQ * NH * SEQ) / 256, 256, 0, stream>>>(kb, kn, BQ * NH * SEQ);

  // 4) fused RBF attention (never materializes the 1 GB score matrix)
  attn_kernel<<<dim3(SEQ / 128, BQ * NH), 256, 0, stream>>>(
      qb, kb, vb, qn, kn, mask, aob);

  // 5) output projection, fp32 stores into d_out
  gemm_wt<1><<<dim3(MROWS / 128, EMB / 128, 1), 256, 0, stream>>>(
      aob, wob, wob, wob, d_out, d_out, d_out);
}
